// MeanAggregator_6657199309166
// MI455X (gfx1250) — matrix operand order repaired and checked by reading, not compiled, not run on hardware
//
#include <hip/hip_runtime.h>
#include <hip/hip_bf16.h>

#define D 128

typedef _Float16 v16h __attribute__((ext_vector_type(16)));
typedef _Float16 v8h  __attribute__((ext_vector_type(8)));
typedef float    v8f  __attribute__((ext_vector_type(8)));

// ---------------------------------------------------------------------------
// Hardware tanh: gfx1250 has V_TANH_F32 (TRANS op). Fall back to a branch-free
// v_exp_f32 + v_rcp_f32 formulation if the builtin isn't in this clang.
// ---------------------------------------------------------------------------
__device__ __forceinline__ float hw_tanh(float x) {
#if __has_builtin(__builtin_amdgcn_tanhf)
    return __builtin_amdgcn_tanhf(x);
#else
    // tanh(x) = 1 - 2/(e^(2x)+1);  e^(2x) = exp2(2*log2(e)*x)
    float e = __builtin_amdgcn_exp2f(x * 2.8853900817779268f);
    return 1.0f - 2.0f * __builtin_amdgcn_rcpf(e + 1.0f);
#endif
}

// ---------------------------------------------------------------------------
// K1: zero out accumulators (out, degree, row_sum)
// ---------------------------------------------------------------------------
__global__ void zero_kernel(float* __restrict__ out, float* __restrict__ degree,
                            float* __restrict__ row_sum, int N) {
    int i = blockIdx.x * blockDim.x + threadIdx.x;
    long total = (long)N * D;
    if (i < total) out[i] = 0.0f;
    if (i < N) { degree[i] = 0.0f; row_sum[i] = 0.0f; }
}

// ---------------------------------------------------------------------------
// K2: degree[s] += (s==d) ? MASK[ind] : 1   (MASK = {1,1,0,0})
// ---------------------------------------------------------------------------
__global__ void degree_kernel(const int* __restrict__ esrc, const int* __restrict__ edst,
                              float* __restrict__ degree, const int* __restrict__ ind,
                              int E) {
    int e = blockIdx.x * blockDim.x + threadIdx.x;
    if (e >= E) return;
    int s = esrc[e], d = edst[e];
    float val = 1.0f;
    if (s == d) val = (ind[0] < 2) ? 1.0f : 0.0f;
    if (val != 0.0f)
        __hip_atomic_fetch_add(&degree[s], val, __ATOMIC_RELAXED, __HIP_MEMORY_SCOPE_AGENT);
}

// ---------------------------------------------------------------------------
// K3: h = tanh(feat @ W1 + b1) @ W2 + b2   via v_wmma_f32_16x16x32_f16
//
// Block = 128 threads = 4 waves; each wave owns a 16-row tile, block = 64 rows.
// LDS: Wbuf = transposed weights as f16 (32 KB, reused W1 then W2),
//      Tbuf = per-wave 16x128 f16 intermediate (16 KB).  Total 48 KB.
// Fragment layouts per CDNA5 ISA 7.12.2 (wave32):
//   A 16x32 f16 : lane<16 -> row=lane, K = {0..7,16..23}; lane>=16 -> K = {8..15,24..31}
//   B 32x16 f16 : lane&15 -> col, lane>>4 selects K {0..7,16..23} / {8..15,24..31}, as A
//   C/D 16x16 f32: col = lane&15, row = 8*(lane>>4) + vgpr
// ---------------------------------------------------------------------------
__global__ __launch_bounds__(128) void mlp_wmma_kernel(
    const float* __restrict__ feat,
    const float* __restrict__ W1, const float* __restrict__ b1,
    const float* __restrict__ W2, const float* __restrict__ b2,
    float* __restrict__ h, int N) {

    __shared__ _Float16 Wbuf[D * D];        // 32 KB, holds W^T (col-major weights)
    __shared__ _Float16 Tbuf[4][16 * D];    // 16 KB, per-wave tanh intermediate

    const int tid   = threadIdx.x;
    const int lane  = tid & 31;
    const int wave  = tid >> 5;
    const int rowBase = blockIdx.x * 64 + wave * 16;

    const int mrow  = lane & 15;   // A-matrix row within tile
    const int khalf = lane >> 4;   // which K half this lane holds
    const int ccol  = lane & 15;   // C/D: N index
    const int crow0 = (lane >> 4) * 8;

    // ---- stage W1 transposed into LDS as f16: Wbuf[n*D + k] = W1[k*D + n] ----
    for (int i = tid; i < D * D; i += 128) {
        int k = i >> 7, n = i & 127;
        Wbuf[n * D + k] = (_Float16)W1[k * D + n];
    }
    __syncthreads();

    // ---- load A fragments of feat (f32 -> f16), rows clamped for the tail ----
    int arow = rowBase + mrow;
    if (arow >= N) arow = N - 1;
    const float* fr = feat + (size_t)arow * D;

    v16h a[4];
#pragma unroll
    for (int kk = 0; kk < 4; ++kk) {
        int kb = kk * 32 + khalf * 8;
#pragma unroll
        for (int i = 0; i < 8; ++i) {
            a[kk][i]     = (_Float16)fr[kb + i];
            a[kk][8 + i] = (_Float16)fr[kb + 16 + i];
        }
    }

    // ---- GEMM1 across 8 output col-tiles, then bias + tanh -> Tbuf ----
#pragma unroll
    for (int j = 0; j < 8; ++j) {
        v8f acc = {};
#pragma unroll
        for (int kk = 0; kk < 4; ++kk) {
            const _Float16* wc = &Wbuf[(j * 16 + ccol) * D + kk * 32 + khalf * 8];
            v8h blo = *(const v8h*)(wc);
            v8h bhi = *(const v8h*)(wc + 16);
            v16h b;
#pragma unroll
            for (int i = 0; i < 8; ++i) { b[i] = blo[i]; b[8 + i] = bhi[i]; }
            acc = __builtin_amdgcn_wmma_f32_16x16x32_f16(false, a[kk], false, b,
                                                         (short)0, acc, false, false);
        }
        float bias = b1[j * 16 + ccol];
#pragma unroll
        for (int v = 0; v < 8; ++v) {
            float t = hw_tanh(acc[v] + bias);
            Tbuf[wave][(crow0 + v) * D + j * 16 + ccol] = (_Float16)t;
        }
    }

    __syncthreads();                 // everyone done reading W1 from Wbuf
    for (int i = tid; i < D * D; i += 128) {   // restage with W2^T
        int k = i >> 7, n = i & 127;
        Wbuf[n * D + k] = (_Float16)W2[k * D + n];
    }
    __syncthreads();

    // ---- A fragments of tanh tile from LDS (ds_load_b128 pairs) ----
    v16h a2[4];
    const _Float16* tr = &Tbuf[wave][mrow * D];
#pragma unroll
    for (int kk = 0; kk < 4; ++kk) {
        int kb = kk * 32 + khalf * 8;
        v8h lo = *(const v8h*)(tr + kb);
        v8h hi = *(const v8h*)(tr + kb + 16);
#pragma unroll
        for (int i = 0; i < 8; ++i) { a2[kk][i] = lo[i]; a2[kk][8 + i] = hi[i]; }
    }

    // ---- GEMM2 + bias, store h (f32) with row guard ----
#pragma unroll
    for (int j = 0; j < 8; ++j) {
        v8f acc = {};
#pragma unroll
        for (int kk = 0; kk < 4; ++kk) {
            const _Float16* wc = &Wbuf[(j * 16 + ccol) * D + kk * 32 + khalf * 8];
            v8h blo = *(const v8h*)(wc);
            v8h bhi = *(const v8h*)(wc + 16);
            v16h b;
#pragma unroll
            for (int i = 0; i < 8; ++i) { b[i] = blo[i]; b[8 + i] = bhi[i]; }
            acc = __builtin_amdgcn_wmma_f32_16x16x32_f16(false, a2[kk], false, b,
                                                         (short)0, acc, false, false);
        }
        float bias = b2[j * 16 + ccol];
#pragma unroll
        for (int v = 0; v < 8; ++v) {
            int orow = rowBase + crow0 + v;
            if (orow < N)
                h[(size_t)orow * D + j * 16 + ccol] = acc[v] + bias;
        }
    }
}

// ---------------------------------------------------------------------------
// K4: one wave32 per edge:
//   w = degree[dst]*W_edge[e]; row_sum[src] += w; out[src][:] += w*h[dst][:]
// h and out live in the 192 MB L2, so gathers/atomics are L2-resident.
// ---------------------------------------------------------------------------
__global__ __launch_bounds__(256) void aggregate_kernel(
    const int* __restrict__ esrc, const int* __restrict__ edst,
    const float* __restrict__ wedge, const float* __restrict__ degree,
    const float* __restrict__ h, float* __restrict__ out,
    float* __restrict__ row_sum, int E) {

    int gid  = blockIdx.x * blockDim.x + threadIdx.x;
    int e    = gid >> 5;
    int lane = gid & 31;
    if (e >= E) return;

    int s = esrc[e], d = edst[e];
    float w = degree[d] * wedge[e];

    if (lane == 0)
        __hip_atomic_fetch_add(&row_sum[s], w, __ATOMIC_RELAXED, __HIP_MEMORY_SCOPE_AGENT);

    // prefetch next edge's h row into cache (global_prefetch_b8)
    if (e + 1 < E)
        __builtin_prefetch(&h[(size_t)edst[e + 1] * D], 0, 0);

    const float* hd = h + (size_t)d * D;
    float* os = out + (size_t)s * D;
#pragma unroll
    for (int i = 0; i < 4; ++i) {
        int c = lane + 32 * i;
        __hip_atomic_fetch_add(&os[c], w * hd[c], __ATOMIC_RELAXED, __HIP_MEMORY_SCOPE_AGENT);
    }
}

// ---------------------------------------------------------------------------
// K5: out /= (row_sum == 0 ? 1 : row_sum)
// ---------------------------------------------------------------------------
__global__ void finalize_kernel(float* __restrict__ out,
                                const float* __restrict__ row_sum, int N) {
    int i = blockIdx.x * blockDim.x + threadIdx.x;
    if (i >= N * D) return;
    float rs = row_sum[i >> 7];
    rs = (rs == 0.0f) ? 1.0f : rs;
    out[i] /= rs;
}

// ---------------------------------------------------------------------------
extern "C" void kernel_launch(void* const* d_in, const int* in_sizes, int n_in,
                              void* d_out, int out_size, void* d_ws, size_t ws_size,
                              hipStream_t stream) {
    const float* feat   = (const float*)d_in[0];
    const float* W_edge = (const float*)d_in[1];
    const float* W1     = (const float*)d_in[2];
    const float* b1     = (const float*)d_in[3];
    const float* W2     = (const float*)d_in[4];
    const float* b2     = (const float*)d_in[5];
    const int*   esrc   = (const int*)d_in[6];
    const int*   edst   = (const int*)d_in[7];
    const int*   ind    = (const int*)d_in[8];

    const int N = in_sizes[0] / D;
    const int E = in_sizes[1];

    float* out     = (float*)d_out;
    float* degree  = (float*)d_ws;           // N floats
    float* row_sum = degree + N;             // N floats
    float* h       = row_sum + N;            // N*D floats (~51 MB)

    // K1: zero accumulators
    {
        long total = (long)N * D;
        int blocks = (int)((total + 255) / 256);
        zero_kernel<<<blocks, 256, 0, stream>>>(out, degree, row_sum, N);
    }
    // K2: degree scatter-add
    degree_kernel<<<(E + 255) / 256, 256, 0, stream>>>(esrc, edst, degree, ind, E);

    // K3: WMMA MLP -> h
    mlp_wmma_kernel<<<(N + 63) / 64, 128, 0, stream>>>(feat, W1, b1, W2, b2, h, N);

    // K4: edge aggregation (one wave per edge)
    {
        long threads = (long)E * 32;
        int blocks = (int)((threads + 255) / 256);
        aggregate_kernel<<<blocks, 256, 0, stream>>>(esrc, edst, W_edge, degree,
                                                     h, out, row_sum, E);
    }
    // K5: normalize
    {
        long total = (long)N * D;
        int blocks = (int)((total + 255) / 256);
        finalize_kernel<<<blocks, 256, 0, stream>>>(out, row_sum, N);
    }
}
